// SSM_13855564497095
// MI455X (gfx1250) — matrix operand order repaired and checked by reading, not compiled, not run on hardware
//
#include <hip/hip_runtime.h>
#include <hip/hip_bf16.h>
#include <math.h>

// ---------------------------------------------------------------------------
// Problem constants (from reference): B=32, CPD=32, F=1024, D=1024, S=128
// proj rows are indexed m = b*F + f  (M = 32768)
// ---------------------------------------------------------------------------
#define EPSN 1e-8f

typedef __attribute__((ext_vector_type(16))) __bf16 v16bf;
typedef __attribute__((ext_vector_type(8)))  __bf16 v8bf;
typedef __attribute__((ext_vector_type(8)))  float  v8f;
typedef int v4i_vs __attribute__((vector_size(16)));   // matches builtin param type

// ---- CDNA5 async global->LDS staging (guarded; falls back to VGPR bounce) --
#if defined(__has_builtin)
#if __has_builtin(__builtin_amdgcn_global_load_async_to_lds_b128)
#define HAVE_ASYNC_LDS 1
#endif
#endif
#ifndef HAVE_ASYNC_LDS
#define HAVE_ASYNC_LDS 0
#endif

__device__ __forceinline__ void stage16(const void* g, void* l) {
#if HAVE_ASYNC_LDS
  // param0: v4i in addrspace(1) (global), param1: addrspace(3) (LDS)
  __builtin_amdgcn_global_load_async_to_lds_b128(
      (__attribute__((address_space(1))) v4i_vs*)g,
      (__attribute__((address_space(3))) v4i_vs*)l, 0, 0);
#else
  *(uint4*)l = *(const uint4*)g;   // global_load_b128 + ds_store_b128
#endif
}
__device__ __forceinline__ void stage_wait() {
#if HAVE_ASYNC_LDS
#if __has_builtin(__builtin_amdgcn_s_wait_asynccnt)
  __builtin_amdgcn_s_wait_asynccnt(0);
#else
  asm volatile("s_wait_asynccnt 0" ::: "memory");
#endif
#endif
}

__device__ __forceinline__ unsigned short f2bf_bits(float x) {
  unsigned int u = __float_as_uint(x);
  unsigned int r = u + 0x7fffu + ((u >> 16) & 1u);   // round-to-nearest-even
  return (unsigned short)(r >> 16);
}
__device__ __forceinline__ __bf16 bf_from_bits(unsigned short b) {
  union { unsigned short s; __bf16 h; } v; v.s = b; return v.h;
}
__device__ __forceinline__ __bf16 f2bf(float x) { return bf_from_bits(f2bf_bits(x)); }

__device__ __forceinline__ v8f zero8() {
  v8f z;
#pragma unroll
  for (int i = 0; i < 8; ++i) z[i] = 0.f;
  return z;
}
__device__ __forceinline__ v16bf cat8(v8bf lo, v8bf hi) {
  v16bf r;
#pragma unroll
  for (int i = 0; i < 8; ++i) { r[i] = lo[i]; r[i + 8] = hi[i]; }
  return r;
}
// 16 contiguous bf16 (ushort storage, 32B) -> v16bf  (two b128 loads)
__device__ __forceinline__ v16bf load16bf(const unsigned short* p) {
  v8bf lo = *reinterpret_cast<const v8bf*>(p);
  v8bf hi = *reinterpret_cast<const v8bf*>(p + 8);
  return cat8(lo, hi);
}

// Butterfly-sum p[0..7] across each 16-lane half of the wave32 (xor 1,2,4,8).
// Matches C/D WMMA layout: lanes 0-15 carry rows 0-7, lanes 16-31 rows 8-15.
__device__ __forceinline__ void rowReduce8(float p[8]) {
#pragma unroll
  for (int m = 1; m < 16; m <<= 1) {
#pragma unroll
    for (int i = 0; i < 8; ++i) p[i] += __shfl_xor(p[i], m, 32);
  }
}

// ---------------------------------------------------------------------------
// K0: fp32 [R][C] -> bf16 transposed [C][R]  (weights become B-operand friendly)
// ---------------------------------------------------------------------------
__global__ void conv_t_kernel(const float* __restrict__ src,
                              unsigned short* __restrict__ dst, int R, int C) {
  int total = R * C;
  for (int idx = blockIdx.x * blockDim.x + threadIdx.x; idx < total;
       idx += gridDim.x * blockDim.x) {
    int r = idx / C, c = idx - r * C;
    dst[c * R + r] = f2bf_bits(src[idx]);
  }
}

// ---------------------------------------------------------------------------
// K1: proj = PLN(ctrl, proj_w).  One 16x16x32 bf16 WMMA per N-tile (K=32 fits
// a single WMMA).  Emits proj bf16 [M][1024] + clamped row norms.
// grid = M/16 blocks x 256 threads (8 waves, wave w owns N-tiles w*8..w*8+7)
// ---------------------------------------------------------------------------
__global__ void proj_kernel(const float* __restrict__ control,
                            const unsigned short* __restrict__ projw_t, // [1024][32]
                            unsigned short* __restrict__ proj_bf16,     // [32768][1024]
                            float* __restrict__ proj_norm) {            // [32768]
  const int lane = threadIdx.x & 31;
  const int w    = threadIdx.x >> 5;
  const int m0   = blockIdx.x * 16;
  const int row16 = lane & 15;
  const int lo = (lane < 16) ? 0 : 8;   // unused; A-frag uses kb too (k = 16h + s)
  const int kb = (lane < 16) ? 0 : 16;  // B-frag K-half select

  const int m = m0 + row16;
  const int b = m >> 10;                // F = 1024
  const int f = m & 1023;

  // Gather A row: ctrl[b, f, k] = control[(b*32 + k)*1024 + f]
  float av[16];
  const float* cb = control + ((size_t)b * 32) * 1024 + f;
#pragma unroll
  for (int j = 0; j < 8; ++j) {
    av[j]     = cb[(size_t)(kb + j)      * 1024];
    av[j + 8] = cb[(size_t)(kb + 8 + j) * 1024];
  }
  float vsq = 0.f;
#pragma unroll
  for (int j = 0; j < 16; ++j) vsq += av[j] * av[j];
  vsq += __shfl_xor(vsq, 16, 32);       // lanes L and L^16 hold the two K-halves

  v16bf a;
#pragma unroll
  for (int j = 0; j < 16; ++j) a[j] = f2bf(av[j]);

  v8f acc[8];
#pragma unroll
  for (int t = 0; t < 8; ++t) {
    const int col = (w * 8 + t) * 16 + row16;
    v16bf bm = load16bf(projw_t + (size_t)col * 32 + kb);
    acc[t] = __builtin_amdgcn_wmma_f32_16x16x32_bf16(false, a, false, bm,
                                                     (short)0, zero8(), false, false);
  }

  __shared__ float partS[8][16];
  __shared__ float vnS[16];
  __shared__ float scS[16];
  float p[8];
#pragma unroll
  for (int i = 0; i < 8; ++i) {
    float s = 0.f;
#pragma unroll
    for (int t = 0; t < 8; ++t) s += acc[t][i] * acc[t][i];
    p[i] = s;
  }
  rowReduce8(p);
  const int base = (lane < 16) ? 0 : 8;
  if ((lane & 15) == 0) {
#pragma unroll
    for (int i = 0; i < 8; ++i) partS[w][base + i] = p[i];
  }
  if (w == 0 && lane < 16) vnS[lane] = vsq;
  __syncthreads();
  if (threadIdx.x < 16) {
    float s = 0.f;
#pragma unroll
    for (int ww = 0; ww < 8; ++ww) s += partS[ww][threadIdx.x];
    float xn = sqrtf(s);
    float vn = sqrtf(vnS[threadIdx.x]);
    float cl = fminf(xn, vn);
    scS[threadIdx.x] = cl / (xn + EPSN);
    proj_norm[m0 + threadIdx.x] = cl;
  }
  __syncthreads();
#pragma unroll
  for (int t = 0; t < 8; ++t) {
    const int col = (w * 8 + t) * 16 + row16;
#pragma unroll
    for (int i = 0; i < 8; ++i) {
      float v = acc[t][i] * scS[base + i];
      proj_bf16[(size_t)(m0 + base + i) * 1024 + col] = f2bf_bits(v);
    }
  }
}

// ---------------------------------------------------------------------------
// K2a: narrow PLN GEMM (N=128): X = A_bf16[M][K] @ W (Wt bf16 [128][K]).
// 16 rows x 256 threads (8 waves, one 16x16 N-tile each).
// ---------------------------------------------------------------------------
__global__ void gemm_pln_n128_kernel(const unsigned short* __restrict__ A,
                                     const unsigned short* __restrict__ Wt,
                                     const float* __restrict__ orig_norm,
                                     float* __restrict__ dst,
                                     int K) {
  const int lane = threadIdx.x & 31;
  const int w    = threadIdx.x >> 5;
  const int m0   = blockIdx.x * 16;
  const int row16 = lane & 15;
  const int lo = (lane < 16) ? 0 : 8;
  const int kb = (lane < 16) ? 0 : 16;

  v8f acc = zero8();
  const unsigned short* arow = A + (size_t)(m0 + row16) * K;
  const int col = w * 16 + row16;

  for (int k0 = 0; k0 < K; k0 += 32) {
    if (k0 + 32 < K) __builtin_prefetch(arow + k0 + 32, 0, 0);
    v16bf a = cat8(*reinterpret_cast<const v8bf*>(arow + k0 + kb),
                   *reinterpret_cast<const v8bf*>(arow + k0 + kb + 8));
    v16bf bm = load16bf(Wt + (size_t)col * K + k0 + kb);
    acc = __builtin_amdgcn_wmma_f32_16x16x32_bf16(false, a, false, bm,
                                                  (short)0, acc, false, false);
  }

  __shared__ float partS[8][16];
  __shared__ float scS[16];
  float p[8];
#pragma unroll
  for (int i = 0; i < 8; ++i) p[i] = acc[i] * acc[i];
  rowReduce8(p);
  const int base = (lane < 16) ? 0 : 8;
  if ((lane & 15) == 0) {
#pragma unroll
    for (int i = 0; i < 8; ++i) partS[w][base + i] = p[i];
  }
  __syncthreads();
  if (threadIdx.x < 16) {
    float s = 0.f;
#pragma unroll
    for (int ww = 0; ww < 8; ++ww) s += partS[ww][threadIdx.x];
    float xn = sqrtf(s);
    float on = orig_norm[m0 + threadIdx.x];
    scS[threadIdx.x] = fminf(xn, on) / (xn + EPSN);
  }
  __syncthreads();
#pragma unroll
  for (int i = 0; i < 8; ++i) {
    dst[(size_t)(m0 + base + i) * 128 + col] = acc[i] * scS[base + i];
  }
}

// ---------------------------------------------------------------------------
// K2b: wide PLN GEMM (N=1024): 32-row M-tile, 512 threads (16 waves x 4
// N-tiles).  A tile (32x32 bf16, 2 KB) is double-buffered in LDS via async
// global->LDS copies (ASYNCcnt + s_wait_asynccnt when available); every B
// fragment feeds two WMMAs (one per row-tile), halving Wt traffic.
// ---------------------------------------------------------------------------
template <bool ACC>
__global__ void __launch_bounds__(512)
gemm_pln32_kernel(const unsigned short* __restrict__ A,
                  const unsigned short* __restrict__ Wt,  // [1024][K]
                  const float* __restrict__ orig_norm,
                  float* __restrict__ dst,                // [M][1024]
                  int K) {
  __shared__ unsigned short ldsA[2][32 * 32];  // 2 x 2 KB double buffer
  __shared__ float partS[16][32];
  __shared__ float scS[32];

  const int tid  = threadIdx.x;
  const int lane = tid & 31;
  const int w    = tid >> 5;
  const int m0   = blockIdx.x * 32;
  const int row16 = lane & 15;
  const int lo = (lane < 16) ? 0 : 8;
  const int kb = (lane < 16) ? 0 : 16;
  const int base = (lane < 16) ? 0 : 8;

  v8f acc[2][4];
#pragma unroll
  for (int mt = 0; mt < 2; ++mt)
#pragma unroll
    for (int t = 0; t < 4; ++t) acc[mt][t] = zero8();

  // stage first A tile: 32 rows x 64 B = 128 x b128, threads 0..127
  const int srow = tid >> 2, sseg = tid & 3;
  if (tid < 128) {
    stage16(A + (size_t)(m0 + srow) * K + sseg * 8,
            &ldsA[0][srow * 32 + sseg * 8]);
  }
  stage_wait();
  __syncthreads();

  int cur = 0;
  for (int k0 = 0; k0 < K; k0 += 32) {
    // prefetch next A tile into the other buffer (async, no wait yet)
    if (tid < 128 && k0 + 32 < K) {
      stage16(A + (size_t)(m0 + srow) * K + (k0 + 32) + sseg * 8,
              &ldsA[cur ^ 1][srow * 32 + sseg * 8]);
    }
    // A fragments from LDS (shared by all 16 waves)
    v16bf a[2];
#pragma unroll
    for (int mt = 0; mt < 2; ++mt) {
      const unsigned short* ar = &ldsA[cur][(mt * 16 + row16) * 32];
      a[mt] = cat8(*reinterpret_cast<const v8bf*>(ar + kb),
                   *reinterpret_cast<const v8bf*>(ar + kb + 8));
    }
#pragma unroll
    for (int t = 0; t < 4; ++t) {
      const int col = (w * 4 + t) * 16 + row16;
      const unsigned short* bp = Wt + (size_t)col * K + k0 + kb;
      if (k0 + 32 < K) __builtin_prefetch(bp + 32, 0, 0);
      v16bf bm = load16bf(bp);
      acc[0][t] = __builtin_amdgcn_wmma_f32_16x16x32_bf16(false, a[0], false, bm,
                                                          (short)0, acc[0][t], false, false);
      acc[1][t] = __builtin_amdgcn_wmma_f32_16x16x32_bf16(false, a[1], false, bm,
                                                          (short)0, acc[1][t], false, false);
    }
    stage_wait();     // next buffer fully landed in LDS
    __syncthreads();  // everyone done reading cur; staging visible to all
    cur ^= 1;
  }

  // per-row ||x||^2 across N=1024 (butterfly within wave, LDS across 16 waves)
#pragma unroll
  for (int mt = 0; mt < 2; ++mt) {
    float p[8];
#pragma unroll
    for (int i = 0; i < 8; ++i) {
      float s = 0.f;
#pragma unroll
      for (int t = 0; t < 4; ++t) s += acc[mt][t][i] * acc[mt][t][i];
      p[i] = s;
    }
    rowReduce8(p);
    if ((lane & 15) == 0) {
#pragma unroll
      for (int i = 0; i < 8; ++i) partS[w][mt * 16 + base + i] = p[i];
    }
  }
  __syncthreads();
  if (tid < 32) {
    float s = 0.f;
#pragma unroll
    for (int ww = 0; ww < 16; ++ww) s += partS[ww][tid];
    float xn = sqrtf(s);
    float on = orig_norm[m0 + tid];
    scS[tid] = fminf(xn, on) / (xn + EPSN);
  }
  __syncthreads();
#pragma unroll
  for (int mt = 0; mt < 2; ++mt) {
#pragma unroll
    for (int t = 0; t < 4; ++t) {
      const int col = (w * 4 + t) * 16 + row16;
#pragma unroll
      for (int i = 0; i < 8; ++i) {
        size_t off = (size_t)(m0 + mt * 16 + base + i) * 1024 + col;
        float v = acc[mt][t][i] * scS[mt * 16 + base + i];
        if (ACC) dst[off] += v; else dst[off] = v;
      }
    }
  }
}

// ---------------------------------------------------------------------------
// K3: sequential scan:  s' = PLN(s, state_matrix);  s <- s' + b_f.
// Single persistent workgroup; state_matrix^T resident in LDS.  Emits s'_f
// bf16 rows (m=b*F+f) + norms so c = PLN(s', output_matrix) batches later.
// ---------------------------------------------------------------------------
__global__ void __launch_bounds__(256, 1)
scan_kernel(const unsigned short* __restrict__ At,      // state_matrix^T bf16 [128][128]
            const float* __restrict__ b_state,          // [32768][128]
            unsigned short* __restrict__ sP_bf16,       // [32768][128]
            float* __restrict__ sP_norm) {              // [32768]
  __shared__ unsigned short AtS[128 * 128];             // 32 KB
  __shared__ float st[32 * 128];                        // 16 KB
  __shared__ float sp[32 * 128];                        // 16 KB
  __shared__ float xpart[8][2][16];
  __shared__ float ssqS[32];
  __shared__ float scS[32];

  const int tid  = threadIdx.x;
  const int lane = tid & 31;
  const int w    = tid >> 5;
  const int row16 = lane & 15;
  const int lo = (lane < 16) ? 0 : 8;
  const int kb = (lane < 16) ? 0 : 16;
  const int base = (lane < 16) ? 0 : 8;

  for (int i = tid; i < 128 * 128; i += 256) AtS[i] = At[i];
  for (int i = tid; i < 32 * 128; i += 256)  st[i] = 0.f;
  __syncthreads();

  for (int f = 0; f < 1024; ++f) {
    if (tid < 32) ssqS[tid] = 0.f;
    __syncthreads();

    // ||state row||^2 via LDS float atomics (ds_add_f32)
    {
      const int r  = tid >> 3;
      const int c0 = (tid & 7) * 16;
      float s = 0.f;
#pragma unroll
      for (int j = 0; j < 16; ++j) { float x = st[r * 128 + c0 + j]; s += x * x; }
      atomicAdd(&ssqS[r], s);
    }

    // x = s @ state_matrix  (M=32 -> 2 m-tiles, N=128 -> tile per wave, K=128)
    v8f acc[2];
#pragma unroll
    for (int mt = 0; mt < 2; ++mt) {
      v8f c = zero8();
      const int row = mt * 16 + row16;
      const int col = w * 16 + row16;
#pragma unroll
      for (int kc = 0; kc < 128; kc += 32) {
        v16bf a;
#pragma unroll
        for (int j = 0; j < 8; ++j) {
          a[j]     = f2bf(st[row * 128 + kc + kb + j]);
          a[j + 8] = f2bf(st[row * 128 + kc + kb + 8 + j]);
        }
        v16bf bm = load16bf(&AtS[col * 128 + kc + kb]);
        c = __builtin_amdgcn_wmma_f32_16x16x32_bf16(false, a, false, bm,
                                                    (short)0, c, false, false);
      }
      acc[mt] = c;
      float p[8];
#pragma unroll
      for (int i = 0; i < 8; ++i) p[i] = c[i] * c[i];
      rowReduce8(p);
      if ((lane & 15) == 0) {
#pragma unroll
        for (int i = 0; i < 8; ++i) xpart[w][mt][base + i] = p[i];
      }
    }
    __syncthreads();

    if (tid < 32) {
      const int mt = tid >> 4, rr = tid & 15;
      float s = 0.f;
#pragma unroll
      for (int ww = 0; ww < 8; ++ww) s += xpart[ww][mt][rr];
      float xn = sqrtf(s);
      float sn = sqrtf(ssqS[tid]);
      float cl = fminf(xn, sn);
      scS[tid] = cl / (xn + EPSN);
      sP_norm[(size_t)tid * 1024 + f] = cl;     // m = b*F + f, b = tid
    }
    __syncthreads();

    // scaled s' -> LDS
#pragma unroll
    for (int mt = 0; mt < 2; ++mt) {
      const int col = w * 16 + row16;
#pragma unroll
      for (int i = 0; i < 8; ++i) {
        const int r = mt * 16 + base + i;
        sp[r * 128 + col] = acc[mt][i] * scS[r];
      }
    }
    __syncthreads();

    // emit s' bf16 and update state: s = s' + b_f   (coalesced 16-elem chunks)
    {
      const int e0 = tid * 16;
      const int bb = e0 >> 7;
      const int k  = e0 & 127;
      const size_t m = (size_t)bb * 1024 + f;
      const float* bp = b_state + m * 128 + k;
      unsigned short* op = sP_bf16 + m * 128 + k;
#pragma unroll
      for (int j = 0; j < 16; ++j) {
        float v = sp[e0 + j];
        op[j] = f2bf_bits(v);
        st[e0 + j] = v + bp[j];
      }
    }
    __syncthreads();
  }
}

// ---------------------------------------------------------------------------
// K4: windowed overlap-add (hop = 512): <=2 frame contributions per sample.
// ---------------------------------------------------------------------------
__global__ void ola_kernel(const float* __restrict__ frames,  // [32768][1024]
                           float* __restrict__ out) {
  const float k2pi = 6.2831853071795864769f / 1024.f;
  const int total = 32 * 524288;
  for (int idx = blockIdx.x * blockDim.x + threadIdx.x; idx < total;
       idx += gridDim.x * blockDim.x) {
    const int b  = idx >> 19;
    const int t  = idx & 524287;
    const int f1 = t >> 9;
    const int n1 = t & 511;
    float w1 = 0.5f * (1.f - __cosf((float)n1 * k2pi));
    float v = frames[(((size_t)b << 10) + f1) * 1024 + n1] * w1;
    if (f1 > 0) {
      const int n2 = n1 + 512;
      float w2 = 0.5f * (1.f - __cosf((float)n2 * k2pi));
      v += frames[(((size_t)b << 10) + f1 - 1) * 1024 + n2] * w2;
    }
    out[idx] = v;
  }
}

// ---------------------------------------------------------------------------
extern "C" void kernel_launch(void* const* d_in, const int* in_sizes, int n_in,
                              void* d_out, int out_size, void* d_ws, size_t ws_size,
                              hipStream_t stream) {
  (void)in_sizes; (void)n_in; (void)out_size; (void)ws_size;
  const float* control  = (const float*)d_in[0];   // [32][32][1024]
  const float* proj_w   = (const float*)d_in[1];   // [32][1024]
  const float* state_m  = (const float*)d_in[2];   // [128][128]
  const float* input_m  = (const float*)d_in[3];   // [1024][128]
  const float* output_m = (const float*)d_in[4];   // [128][1024]
  const float* direct_m = (const float*)d_in[5];   // [1024][1024]
  float* out = (float*)d_out;

  char* ws = (char*)d_ws;
  size_t off = 0;
  auto take = [&](size_t bytes) -> char* {
    char* p = ws + off;
    off = (off + bytes + 255) & ~(size_t)255;
    return p;
  };
  unsigned short* proj_bf   = (unsigned short*)take((size_t)32768 * 1024 * 2); // 64 MB
  float*          proj_norm = (float*)take((size_t)32768 * 4);
  float*          b_state   = (float*)take((size_t)32768 * 128 * 4);           // 16 MB
  float*          frames    = (float*)take((size_t)32768 * 1024 * 4);          // 128 MB
  unsigned short* sP_bf     = (unsigned short*)take((size_t)32768 * 128 * 2);  // 8 MB
  float*          sP_norm   = (float*)take((size_t)32768 * 4);
  unsigned short* projw_t   = (unsigned short*)take((size_t)32 * 1024 * 2);
  unsigned short* inputm_t  = (unsigned short*)take((size_t)1024 * 128 * 2);
  unsigned short* outputm_t = (unsigned short*)take((size_t)128 * 1024 * 2);
  unsigned short* directm_t = (unsigned short*)take((size_t)1024 * 1024 * 2);
  unsigned short* statem_t  = (unsigned short*)take((size_t)128 * 128 * 2);

  // bf16 transposed weight copies (B-operand layout: Wt[n][k])
  conv_t_kernel<<<64,   256, 0, stream>>>(proj_w,   projw_t,   32,   1024);
  conv_t_kernel<<<256,  256, 0, stream>>>(input_m,  inputm_t,  1024, 128);
  conv_t_kernel<<<256,  256, 0, stream>>>(output_m, outputm_t, 128,  1024);
  conv_t_kernel<<<2048, 256, 0, stream>>>(direct_m, directm_t, 1024, 1024);
  conv_t_kernel<<<64,   256, 0, stream>>>(state_m,  statem_t,  128,  128);

  // proj = PLN(ctrl, proj_w)
  proj_kernel<<<2048, 256, 0, stream>>>(control, projw_t, proj_bf, proj_norm);
  // b = PLN(proj, input_matrix)           [32768,1024]@[1024,128]
  gemm_pln_n128_kernel<<<2048, 256, 0, stream>>>(proj_bf, inputm_t, proj_norm,
                                                 b_state, 1024);
  // frames = d = PLN(proj, direct_matrix) [32768,1024]@[1024,1024]
  gemm_pln32_kernel<false><<<1024, 512, 0, stream>>>(proj_bf, directm_t, proj_norm,
                                                     frames, 1024);
  // sequential recurrence -> s'_f (bf16) + norms
  scan_kernel<<<1, 256, 0, stream>>>(statem_t, b_state, sP_bf, sP_norm);
  // frames += c = PLN(s', output_matrix)  [32768,128]@[128,1024]
  gemm_pln32_kernel<true><<<1024, 512, 0, stream>>>(sP_bf, outputm_t, sP_norm,
                                                    frames, 128);
  // windowed overlap-add
  ola_kernel<<<8192, 256, 0, stream>>>(frames, out);
}
